// BDDC_2D_79474074845310
// MI455X (gfx1250) — matrix operand order repaired and checked by reading, not compiled, not run on hardware
//
#include <hip/hip_runtime.h>

#define MM 384
#define CC 16
#define PADW 392                       // 384 + 2*4 halo
#define PPITCH (PADW*CC)               // u16 elems per padded row
#define NPIX (MM*MM)
#define NEL  (NPIX*CC)                 // 2359296 per plane
#define NPAD (PADW*PADW*CC)            // padded bf16 elems
#define NPAIR 41                       // 81 taps -> 41 pairs (last half zero)
#define WSLICE (3*NPAIR*32*16)         // u16 per direction: {Re,Im,-Im} fragments

typedef unsigned short u16;
typedef __attribute__((ext_vector_type(16))) __bf16 v16bf;
typedef __attribute__((ext_vector_type(8)))  __bf16 v8bf;
typedef __attribute__((ext_vector_type(8)))  float  v8f;

__device__ __forceinline__ u16 f2bf(float f) {
  unsigned int u = __float_as_uint(f);
  return (u16)((u + 0x7FFFu + ((u >> 16) & 1u)) >> 16);
}

#define WMMA_BF16(A, B, C) __builtin_amdgcn_wmma_f32_16x16x32_bf16(false, A, false, B, (short)0, C, false, false)

// ---------------- prep: zero padded bf16 buffers (halo stays 0) ----------------
__global__ __launch_bounds__(256) void k_clear(u16* a, u16* b, u16* c, u16* d) {
  int st = gridDim.x * blockDim.x;
  for (int i = blockIdx.x * blockDim.x + threadIdx.x; i < NPAD; i += st) {
    a[i] = 0; b[i] = 0; c[i] = 0; d[i] = 0;
  }
}

// pack ws (= prev_kspace, interleaved [pix*32 + c | +16]) into padded bf16 planes
__global__ __launch_bounds__(256) void k_pack_ws(const float* __restrict__ pk,
                                                 u16* __restrict__ pbr, u16* __restrict__ pbi) {
  int st = gridDim.x * blockDim.x;
  for (int i = blockIdx.x * blockDim.x + threadIdx.x; i < NEL; i += st) {
    int pix = i >> 4, c = i & 15, y = pix / MM, x = pix % MM;
    int gi = pix * 32 + c;
    int po = (y + 4) * PPITCH + (x + 4) * CC + c;
    pbr[po] = f2bf(pk[gi]);
    pbi[po] = f2bf(pk[gi + 16]);
  }
}

// Build WMMA B-fragments for Wf (dir0) and Wb (dir1), comps {Re, Im, -Im}.
// Wf[co,ci,ky,kx] =  K[kx,ky,ci,co]          (kernel flat: (kx*9+ky)*256 + ci*16 + co)
// Wb[co,ci,ky,kx] = conj(K[8-kx,8-ky,co,ci]) (flat: ((8-kx)*9+(8-ky))*256 + co*16 + ci)
// B layout: lane l -> (co=l&15, h=l>>4); element j -> tap = 2*pair+(j>>3), ci=h*8+(j&7)
__global__ __launch_bounds__(256) void k_prep_w(const float* __restrict__ kr,
                                                const float* __restrict__ ki,
                                                u16* __restrict__ wbuf) {
  int tot = 2 * WSLICE;
  int st = gridDim.x * blockDim.x;
  for (int idx = blockIdx.x * blockDim.x + threadIdx.x; idx < tot; idx += st) {
    int j    = idx & 15;
    int lane = (idx >> 4) & 31;
    int pr   = (idx >> 9) % NPAIR;
    int comp = (idx / (512 * NPAIR)) % 3;
    int dir  = idx / WSLICE;
    int co = lane & 15, h = lane >> 4;
    int tap = 2 * pr + (j >> 3), ci = h * 8 + (j & 7);
    float wr = 0.f, wi = 0.f;
    if (tap <= 80) {
      int ky = tap / 9, kx = tap % 9;
      if (dir == 0) {
        int f = (kx * 9 + ky) * 256 + ci * 16 + co;
        wr = kr[f]; wi = ki[f];
      } else {
        int f = ((8 - kx) * 9 + (8 - ky)) * 256 + co * 16 + ci;
        wr = kr[f]; wi = -ki[f];
      }
    }
    float v = (comp == 0) ? wr : (comp == 1) ? wi : -wi;
    wbuf[idx] = f2bf(v);
  }
}

// ---------------- the WMMA complex conv: D = G(x,W), fused epilogues ----------------
// Each wave computes a 16(x) x 2(y) pixel strip, all 16 output channels, so the three
// B-weight fragments per tap-pair are amortized over 8 WMMAs.
// mode 0: t = G(x)-x          -> outr/i (f32) + obr/i (bf16 padded)
// mode 1: q = (G(t)-t) + mask(p) + am*p              (ar_/ai_ = p planes)
// mode 2: p0 = mask(kk) + am*z - (G(t)-t) - mask(ws) - am*ws
__global__ __launch_bounds__(256)
void k_conv(const u16* __restrict__ xbr, const u16* __restrict__ xbi,
            const u16* __restrict__ wt,
            const float* __restrict__ sr, const float* __restrict__ si, int s_ilv,
            float* __restrict__ outr, float* __restrict__ outi,
            u16* __restrict__ obr, u16* __restrict__ obi,
            const float* __restrict__ ar_, const float* __restrict__ ai_,
            const float* __restrict__ ksp, const float* __restrict__ zilv,
            const float* __restrict__ wilv, const float* __restrict__ miu, int mode) {
  const int lane = threadIdx.x & 31;
  const int wave = threadIdx.x >> 5;
  const int wid  = blockIdx.x * 8 + wave;     // 4608 wave-strips
  const int y0 = (wid / 24) * 2;
  const int x0 = (wid % 24) * 16;
  const int m  = lane & 15;
  const int h  = lane >> 4;
  const int colbase = (x0 + m) * CC + h * 8;

  v8f dr0 = {}, di0 = {}, dr1 = {}, di1 = {};
  v8bf zro;
#pragma unroll
  for (int k = 0; k < 8; ++k) zro[k] = (__bf16)0.0f;

  // loop-invariant per-lane bases: all tap offsets below are compile-time immediates
  const u16* xr0 = xbr + y0 * PPITCH + colbase;
  const u16* xi0 = xbi + y0 * PPITCH + colbase;
  const u16* wtb = wt + lane * 16;

#pragma unroll
  for (int p = 0; p < NPAIR; ++p) {
    const int t0 = 2 * p, t1 = 2 * p + 1;
    const int o0 = (t0 / 9) * PPITCH + (t0 % 9) * CC;
    v8bf r0a = *(const v8bf*)(xr0 + o0);
    v8bf i0a = *(const v8bf*)(xi0 + o0);
    v8bf r1a = *(const v8bf*)(xr0 + PPITCH + o0);
    v8bf i1a = *(const v8bf*)(xi0 + PPITCH + o0);
    v8bf r0b = zro, i0b = zro, r1b = zro, i1b = zro;
    if (t1 <= 80) {
      const int o1 = (t1 / 9) * PPITCH + (t1 % 9) * CC;
      r0b = *(const v8bf*)(xr0 + o1);
      i0b = *(const v8bf*)(xi0 + o1);
      r1b = *(const v8bf*)(xr0 + PPITCH + o1);
      i1b = *(const v8bf*)(xi0 + PPITCH + o1);
    }
    v16bf A0r = __builtin_shufflevector(r0a, r0b, 0,1,2,3,4,5,6,7,8,9,10,11,12,13,14,15);
    v16bf A0i = __builtin_shufflevector(i0a, i0b, 0,1,2,3,4,5,6,7,8,9,10,11,12,13,14,15);
    v16bf A1r = __builtin_shufflevector(r1a, r1b, 0,1,2,3,4,5,6,7,8,9,10,11,12,13,14,15);
    v16bf A1i = __builtin_shufflevector(i1a, i1b, 0,1,2,3,4,5,6,7,8,9,10,11,12,13,14,15);
    const v16bf br  = *(const v16bf*)(wtb + (0 * NPAIR + p) * 512);
    const v16bf bi  = *(const v16bf*)(wtb + (1 * NPAIR + p) * 512);
    const v16bf bin = *(const v16bf*)(wtb + (2 * NPAIR + p) * 512);
    // Dr += Ar*Wr + Ai*(-Wi);  Di += Ar*Wi + Ai*Wr   (both rows share B fragments)
    dr0 = WMMA_BF16(A0r, br,  dr0);
    dr0 = WMMA_BF16(A0i, bin, dr0);
    di0 = WMMA_BF16(A0r, bi,  di0);
    di0 = WMMA_BF16(A0i, br,  di0);
    dr1 = WMMA_BF16(A1r, br,  dr1);
    dr1 = WMMA_BF16(A1i, bin, dr1);
    di1 = WMMA_BF16(A1r, bi,  di1);
    di1 = WMMA_BF16(A1i, br,  di1);
  }

  const float am = fabsf(miu[0]);
  const int co = m;
#pragma unroll
  for (int rw = 0; rw < 2; ++rw) {
    const v8f& DR = rw ? dr1 : dr0;
    const v8f& DI = rw ? di1 : di0;
    const int y = y0 + rw;
#pragma unroll
    for (int v = 0; v < 8; ++v) {
      const int px = x0 + v + h * 8;
      const int g  = (y * MM + px) * CC + co;   // planar
      const int gi = (y * MM + px) * 32 + co;   // interleaved
      const float ssr = s_ilv ? sr[gi] : sr[g];
      const float ssi = s_ilv ? sr[gi + 16] : si[g];
      float vr = DR[v] - ssr;
      float vi = DI[v] - ssi;
      if (mode == 0) {
        outr[g] = vr; outi[g] = vi;
        const int po = (y + 4) * PPITCH + (px + 4) * CC + co;
        obr[po] = f2bf(vr); obi[po] = f2bf(vi);
      } else if (mode == 1) {
        const bool ms = (ksp[gi] != 0.f) || (ksp[gi + 16] != 0.f);
        const float pv = ar_[g], qv = ai_[g];
        outr[g] = vr + (ms ? pv : 0.f) + am * pv;
        outi[g] = vi + (ms ? qv : 0.f) + am * qv;
      } else {
        const bool ms = (ksp[gi] != 0.f) || (ksp[gi + 16] != 0.f);
        const float wr = wilv[gi], wi = wilv[gi + 16];
        outr[g] = (ms ? ksp[gi] : 0.f)      + am * zilv[gi]      - vr - (ms ? wr : 0.f) - am * wr;
        outi[g] = (ms ? ksp[gi + 16] : 0.f) + am * zilv[gi + 16] - vi - (ms ? wi : 0.f) - am * wi;
      }
    }
  }
}

// ---------------- CG glue kernels ----------------
__device__ __forceinline__ float blk_reduce(float v, float* sm) {
  sm[threadIdx.x] = v; __syncthreads();
  for (int s = 128; s > 0; s >>= 1) {
    if (threadIdx.x < (unsigned)s) sm[threadIdx.x] += sm[threadIdx.x + s];
    __syncthreads();
  }
  return sm[0];
}

// r=p, b=ws (into d_out interleaved), pack p->bf16, partial |p|^2
__global__ __launch_bounds__(256)
void k_init2(const float* pr, const float* pi, const float* pk, float* bout,
             float* rr_, float* ri_, u16* pbr, u16* pbi, float* part) {
  __shared__ float sm[256];
  int st = gridDim.x * blockDim.x;
  float acc = 0.f;
  for (int i = blockIdx.x * blockDim.x + threadIdx.x; i < NEL; i += st) {
    int pix = i >> 4, c = i & 15, y = pix / MM, x = pix % MM;
    int gi = pix * 32 + c;
    float a = pr[i], b = pi[i];
    rr_[i] = a; ri_[i] = b;
    bout[gi] = pk[gi]; bout[gi + 16] = pk[gi + 16];
    int po = (y + 4) * PPITCH + (x + 4) * CC + c;
    pbr[po] = f2bf(a); pbi[po] = f2bf(b);
    acc += a * a + b * b;
  }
  float s = blk_reduce(acc, sm);
  if (threadIdx.x == 0) part[blockIdx.x] = s;
}

__global__ __launch_bounds__(256) void k_fin_rr0(const float* part, float* sc) {
  __shared__ float sm[256];
  float s = blk_reduce(part[threadIdx.x], sm);
  if (threadIdx.x == 0) sc[0] = s;          // rr
}

__global__ __launch_bounds__(256)
void k_dot_qp(const float* qr, const float* qi, const float* pr, const float* pi, float* part) {
  __shared__ float sm[256];
  int st = gridDim.x * blockDim.x;
  float a = 0.f, b = 0.f;
  for (int i = blockIdx.x * blockDim.x + threadIdx.x; i < NEL; i += st) {
    float q0 = qr[i], q1 = qi[i], p0 = pr[i], p1 = pi[i];
    a += q0 * p0 + q1 * p1;                 // Re(q * conj(p))
    b += q1 * p0 - q0 * p1;                 // Im(q * conj(p))
  }
  float s0 = blk_reduce(a, sm); __syncthreads();
  float s1 = blk_reduce(b, sm);
  if (threadIdx.x == 0) { part[blockIdx.x] = s0; part[256 + blockIdx.x] = s1; }
}

__global__ __launch_bounds__(256) void k_fin_alpha(const float* part, float* sc) {
  __shared__ float sm[256];
  float re = blk_reduce(part[threadIdx.x], sm); __syncthreads();
  float im = blk_reduce(part[256 + threadIdx.x], sm);
  if (threadIdx.x == 0) {
    float rr = sc[0], den = re * re + im * im;
    sc[1] =  rr * re / den;                 // alpha = rr / (re + i im)
    sc[2] = -rr * im / den;
  }
}

// b += alpha*p (interleaved out), r -= alpha*q, partial |r|^2
__global__ __launch_bounds__(256)
void k_update1(float* bout, float* rr_, float* ri_, const float* pr, const float* pi,
               const float* qr, const float* qi, const float* sc, float* part) {
  __shared__ float sm[256];
  const float are = sc[1], aim = sc[2];
  int st = gridDim.x * blockDim.x;
  float acc = 0.f;
  for (int i = blockIdx.x * blockDim.x + threadIdx.x; i < NEL; i += st) {
    int gi = (i >> 4) * 32 + (i & 15);
    float p0 = pr[i], p1 = pi[i];
    bout[gi]      += are * p0 - aim * p1;
    bout[gi + 16] += are * p1 + aim * p0;
    float r0 = rr_[i] - (are * qr[i] - aim * qi[i]);
    float r1 = ri_[i] - (are * qi[i] + aim * qr[i]);
    rr_[i] = r0; ri_[i] = r1;
    acc += r0 * r0 + r1 * r1;
  }
  float s = blk_reduce(acc, sm);
  if (threadIdx.x == 0) part[blockIdx.x] = s;
}

__global__ __launch_bounds__(256) void k_fin_beta(const float* part, float* sc) {
  __shared__ float sm[256];
  float s = blk_reduce(part[threadIdx.x], sm);
  if (threadIdx.x == 0) { sc[3] = s / sc[0]; sc[0] = s; }   // beta, rr <- rr_new
}

// p = r + beta*p, repack p -> bf16 padded
__global__ __launch_bounds__(256)
void k_update2(float* pr, float* pi, const float* rr_, const float* ri_,
               const float* sc, u16* pbr, u16* pbi) {
  const float beta = sc[3];
  int st = gridDim.x * blockDim.x;
  for (int i = blockIdx.x * blockDim.x + threadIdx.x; i < NEL; i += st) {
    int pix = i >> 4, c = i & 15, y = pix / MM, x = pix % MM;
    float a = rr_[i] + beta * pr[i];
    float b = ri_[i] + beta * pi[i];
    pr[i] = a; pi[i] = b;
    int po = (y + 4) * PPITCH + (x + 4) * CC + c;
    pbr[po] = f2bf(a); pbi[po] = f2bf(b);
  }
}

// ---------------- launch ----------------
extern "C" void kernel_launch(void* const* d_in, const int* in_sizes, int n_in,
                              void* d_out, int out_size, void* d_ws, size_t ws_size,
                              hipStream_t stream) {
  const float* z   = (const float*)d_in[0];
  const float* ksp = (const float*)d_in[1];
  const float* krn_r = (const float*)d_in[2];
  const float* krn_i = (const float*)d_in[3];
  const float* pk  = (const float*)d_in[5];
  const float* miu = (const float*)d_in[6];
  float* bout = (float*)d_out;

  char* w = (char*)d_ws;
  auto alloc = [&](size_t bytes) { char* p = w; w += (bytes + 255) & ~size_t(255); return p; };
  u16* pbr = (u16*)alloc(NPAD * 2);
  u16* pbi = (u16*)alloc(NPAD * 2);
  u16* tbr = (u16*)alloc(NPAD * 2);
  u16* tbi = (u16*)alloc(NPAD * 2);
  u16* wbuf = (u16*)alloc(2 * WSLICE * 2);
  float* pr = (float*)alloc(NEL * 4); float* pi = (float*)alloc(NEL * 4);
  float* rr_ = (float*)alloc(NEL * 4); float* ri_ = (float*)alloc(NEL * 4);
  float* tr = (float*)alloc(NEL * 4); float* ti = (float*)alloc(NEL * 4);
  float* qr = (float*)alloc(NEL * 4); float* qi = (float*)alloc(NEL * 4);
  float* part = (float*)alloc(512 * 4);
  float* sc = (float*)alloc(16 * 4);

  const u16* WF = wbuf;
  const u16* WB = wbuf + WSLICE;
  const int GB = 256, TB = 256, GC = 576;   // 576 blocks * 8 waves * (16x2 px) = 384x384

  k_clear<<<GB, TB, 0, stream>>>(pbr, pbi, tbr, tbi);
  k_pack_ws<<<GB, TB, 0, stream>>>(pk, pbr, pbi);
  k_prep_w<<<(2 * WSLICE + 255) / 256, 256, 0, stream>>>(krn_r, krn_i, wbuf);

  // t = G(ws,Wf) - ws ;  p0 = mask(kk) + am*z - (G(t,Wb)-t) - mask(ws) - am*ws
  k_conv<<<GC, TB, 0, stream>>>(pbr, pbi, WF, pk, nullptr, 1, tr, ti, tbr, tbi,
                                nullptr, nullptr, nullptr, nullptr, nullptr, miu, 0);
  k_conv<<<GC, TB, 0, stream>>>(tbr, tbi, WB, tr, ti, 0, pr, pi, nullptr, nullptr,
                                nullptr, nullptr, ksp, z, pk, miu, 2);
  k_init2<<<GB, TB, 0, stream>>>(pr, pi, pk, bout, rr_, ri_, pbr, pbi, part);
  k_fin_rr0<<<1, TB, 0, stream>>>(part, sc);

  for (int it = 0; it < 10; ++it) {
    k_conv<<<GC, TB, 0, stream>>>(pbr, pbi, WF, pr, pi, 0, tr, ti, tbr, tbi,
                                  nullptr, nullptr, nullptr, nullptr, nullptr, miu, 0);
    k_conv<<<GC, TB, 0, stream>>>(tbr, tbi, WB, tr, ti, 0, qr, qi, nullptr, nullptr,
                                  pr, pi, ksp, nullptr, nullptr, miu, 1);
    k_dot_qp<<<GB, TB, 0, stream>>>(qr, qi, pr, pi, part);
    k_fin_alpha<<<1, TB, 0, stream>>>(part, sc);
    k_update1<<<GB, TB, 0, stream>>>(bout, rr_, ri_, pr, pi, qr, qi, sc, part);
    k_fin_beta<<<1, TB, 0, stream>>>(part, sc);
    k_update2<<<GB, TB, 0, stream>>>(pr, pi, rr_, ri_, sc, pbr, pbi);
  }
}
